// ChaosAttention_26620207301247
// MI455X (gfx1250) — matrix operand order repaired and checked by reading, not compiled, not run on hardware
//
#include <hip/hip_runtime.h>
#include <hip/hip_bf16.h>
#include <math.h>

typedef __attribute__((ext_vector_type(16))) _Float16 v16h;
typedef __attribute__((ext_vector_type(8)))  _Float16 v8h;
typedef __attribute__((ext_vector_type(8)))  float    v8f;
typedef __attribute__((ext_vector_type(4)))  unsigned int v4u;
typedef __attribute__((ext_vector_type(8)))  int      v8i;
typedef __attribute__((ext_vector_type(4)))  int      v4i;

// Problem constants (fixed by the reference)
#define EDIM 1024
#define HHEADS 16
#define DHEAD 64
#define TSEQ 2048
#define BATCH 2
#define NPHYS 2
#define PHID 16

#if __has_builtin(__builtin_amdgcn_tensor_load_to_lds)
#define USE_TDM 1
#else
#define USE_TDM 0
#endif

// ---------------------------------------------------------------------------
// Fragment helpers (contiguous k order used here, 16-bit 16x16x32):
//  A-frag: lane L -> row M = L&15, g = L>>4; elems e0..7 at K = kb + g*16 + e,
//          elems e8..15 at K = kb + g*16 + e  -> two 16B loads.
//  B-frag: lane L -> col N = L&15, g = L>>4; elems e0..15 at K = kb + g*16 + e
//          -> one contiguous 32B load.
// ---------------------------------------------------------------------------
static __device__ inline v16h frag_a_2x8(const _Float16* p0, const _Float16* p1) {
    v8h lo = *(const v8h*)p0;
    v8h hi = *(const v8h*)p1;
    v16h r;
#pragma unroll
    for (int i = 0; i < 8; ++i) { r[i] = lo[i]; r[i + 8] = hi[i]; }
    return r;
}

// ---------------------------------------------------------------------------
// Tensor Data Mover: 2D tile (tile_x elems of 2B per row, tile_y rows) from a
// row-major f16 tensor (row stride = stride0 elems) into contiguous LDS.
// Descriptor packing per CDNA5 ISA.
// ---------------------------------------------------------------------------
#if USE_TDM
static __device__ inline void tdm_load_tile_2d(unsigned int lds_off_bytes,
                                               const _Float16* gptr,
                                               int tile_x, int tile_y,
                                               int tensor_d0, int tensor_d1,
                                               int stride0) {
    unsigned long long ga = (unsigned long long)(uintptr_t)gptr;
    v4u g0;
    g0[0] = 1u;                                        // count=1, user desc, no gather
    g0[1] = lds_off_bytes;                             // lds_addr
    g0[2] = (unsigned int)ga;                          // global_addr[31:0]
    g0[3] = (unsigned int)((ga >> 32) & 0x1FFFFFFu) | (2u << 30); // addr[56:32] | type=2
    v8i g1;
    g1[0] = (1 << 16);                                 // data_size=1 (2 bytes), mask=0
    g1[1] = (tensor_d0 & 0xFFFF) << 16;                // tensor_dim0[15:0]
    g1[2] = ((tensor_d0 >> 16) & 0xFFFF) | ((tensor_d1 & 0xFFFF) << 16);
    g1[3] = ((tensor_d1 >> 16) & 0xFFFF) | ((tile_x & 0xFFFF) << 16); // tile_dim0
    g1[4] = (tile_y & 0xFFFF);                         // tile_dim1 (tile_dim2 = 0)
    g1[5] = stride0;                                   // tensor_dim0_stride[31:0]
    g1[6] = 0;
    g1[7] = 0;
    v4i g2 = {};                                       // tensor_dim2.. (unused, 2D)
    v4i g3 = {};
#if defined(__clang_major__) && (__clang_major__ >= 23)
    v8i gex = {};
    __builtin_amdgcn_tensor_load_to_lds(g0, g1, g2, g3, gex, 0);
#else
    __builtin_amdgcn_tensor_load_to_lds(g0, g1, g2, g3, 0);
#endif
}
#endif

// ---------------------------------------------------------------------------
// Kernel 0a: convert x fp32 -> f16 (contiguous)
// ---------------------------------------------------------------------------
__global__ __launch_bounds__(256) void convert_x_kernel(const float* __restrict__ x,
                                                        _Float16* __restrict__ xh) {
    size_t i = ((size_t)blockIdx.x * 256 + threadIdx.x) * 8;
    float4 a = *(const float4*)&x[i];
    float4 b = *(const float4*)&x[i + 4];
    v8h o;
    o[0] = (_Float16)a.x; o[1] = (_Float16)a.y; o[2] = (_Float16)a.z; o[3] = (_Float16)a.w;
    o[4] = (_Float16)b.x; o[5] = (_Float16)b.y; o[6] = (_Float16)b.z; o[7] = (_Float16)b.w;
    *(v8h*)&xh[i] = o;
}

// ---------------------------------------------------------------------------
// Kernel 0b: convert + transpose weights: Wt[n][k] = (f16) W[k][n]
// blockIdx.y selects which of the 4 weight matrices.
// ---------------------------------------------------------------------------
__global__ __launch_bounds__(256) void convert_w_kernel(
    const float* __restrict__ Wq, const float* __restrict__ Wk,
    const float* __restrict__ Wv, const float* __restrict__ Wo,
    _Float16* __restrict__ Wqh, _Float16* __restrict__ Wkh,
    _Float16* __restrict__ Wvh, _Float16* __restrict__ Woh) {
    const int z = blockIdx.y;
    const float* __restrict__ W = (z == 0) ? Wq : (z == 1) ? Wk : (z == 2) ? Wv : Wo;
    _Float16* __restrict__ Wt  = (z == 0) ? Wqh : (z == 1) ? Wkh : (z == 2) ? Wvh : Woh;
    size_t idx = (size_t)blockIdx.x * 256 + threadIdx.x; // over k-major element order
    int n = (int)(idx & (EDIM - 1));
    int k = (int)(idx >> 10);
    Wt[(size_t)n * EDIM + k] = (_Float16)W[(size_t)k * EDIM + n];
}

// ---------------------------------------------------------------------------
// Kernel 1: physics adapter  bias[b,h] = gelu(ps@Wp1+bp1) @ Wp2 + bp2
// ---------------------------------------------------------------------------
__global__ void physics_bias_kernel(const float* __restrict__ ps,
                                    const float* __restrict__ Wp1,
                                    const float* __restrict__ bp1,
                                    const float* __restrict__ Wp2,
                                    const float* __restrict__ bp2,
                                    float* __restrict__ out) {
    int tid = threadIdx.x;
    if (tid >= BATCH * HHEADS) return;
    int b = tid / HHEADS, h = tid % HHEADS;
    float acc = bp2[h];
#pragma unroll
    for (int j = 0; j < PHID; ++j) {
        float a = bp1[j];
#pragma unroll
        for (int i = 0; i < NPHYS; ++i) a += ps[b * NPHYS + i] * Wp1[i * PHID + j];
        float g = 0.5f * a * (1.0f + erff(a * 0.70710678118654752f)); // exact gelu
        acc += g * Wp2[j * HHEADS + h];
    }
    out[tid] = acc;
}

// ---------------------------------------------------------------------------
// GEMM core used by both projection kernels: 128x128 C tile, K-step 32,
// TDM double-buffered LDS staging (DMA for step kb+32 overlaps WMMA on kb).
// acc is the caller's 2x4 array of 16x16 accumulators.
// ---------------------------------------------------------------------------
#define GEMM_MAIN_LOOP(Aptr, Bptr)                                                  \
    do {                                                                            \
        _Float16(*Asb)[128][32] = As;                                               \
        _Float16(*Bsb)[128][32] = Bs;                                               \
        int cur = 0;                                                                \
        PIPE_PROLOGUE(Aptr, Bptr);                                                  \
        for (int kb = 0; kb < EDIM; kb += 32) {                                     \
            PIPE_ISSUE_NEXT(Aptr, Bptr);                                            \
            __syncthreads();                                                        \
            v16h af[2];                                                             \
            _Pragma("unroll") for (int ms = 0; ms < 2; ++ms) {                      \
                const _Float16* rp = &Asb[cur][wm * 32 + ms * 16 + l16][0];         \
                af[ms] = frag_a_2x8(rp + half * 16, rp + half * 16 + 8);            \
            }                                                                       \
            _Pragma("unroll") for (int ns = 0; ns < 4; ++ns) {                      \
                const v16h bf =                                                     \
                    *(const v16h*)&Bsb[cur][wn * 64 + ns * 16 + l16][half * 16];    \
                _Pragma("unroll") for (int ms = 0; ms < 2; ++ms)                    \
                    acc[ms][ns] = __builtin_amdgcn_wmma_f32_16x16x32_f16(           \
                        false, af[ms], false, bf, (short)0, acc[ms][ns], false,     \
                        false);                                                     \
            }                                                                       \
            __syncthreads();                                                        \
            cur ^= 1;                                                               \
        }                                                                           \
    } while (0)

#if USE_TDM
#define PIPE_PROLOGUE(Aptr, Bptr)                                                   \
    if (wid == 0) {                                                                 \
        tdm_load_tile_2d((unsigned int)(uintptr_t)&As[0][0][0], (Aptr), 32, 128,    \
                         EDIM, BATCH * TSEQ, EDIM);                                 \
        tdm_load_tile_2d((unsigned int)(uintptr_t)&Bs[0][0][0], (Bptr), 32, 128,    \
                         EDIM, EDIM, EDIM);                                         \
    }
// Issue next K-step into the other buffer (dummy re-load of kb=0 on last step
// keeps the in-order TENSORcnt<=2 invariant without a dynamic wait immediate).
#define PIPE_ISSUE_NEXT(Aptr, Bptr)                                                 \
    if (wid == 0) {                                                                 \
        const int kbn = (kb + 32 < EDIM) ? (kb + 32) : 0;                           \
        tdm_load_tile_2d((unsigned int)(uintptr_t)&As[cur ^ 1][0][0], (Aptr) + kbn, \
                         32, 128, EDIM, BATCH * TSEQ, EDIM);                        \
        tdm_load_tile_2d((unsigned int)(uintptr_t)&Bs[cur ^ 1][0][0], (Bptr) + kbn, \
                         32, 128, EDIM, EDIM, EDIM);                                \
        __builtin_amdgcn_s_wait_tensorcnt(2);                                       \
    }
#else
#define PIPE_PROLOGUE(Aptr, Bptr)
#define PIPE_ISSUE_NEXT(Aptr, Bptr)                                                 \
    {                                                                               \
        const int c8 = (tid & 3) * 8;                                               \
        const int r0 = tid >> 2;                                                    \
        _Pragma("unroll") for (int i = 0; i < 2; ++i) {                             \
            int row = r0 + 64 * i;                                                  \
            *(v8h*)&As[cur][row][c8] =                                              \
                *(const v8h*)&(Aptr)[(size_t)row * EDIM + kb + c8];                 \
            *(v8h*)&Bs[cur][row][c8] =                                              \
                *(const v8h*)&(Bptr)[(size_t)row * EDIM + kb + c8];                 \
        }                                                                           \
    }
#endif

// ---------------------------------------------------------------------------
// Kernel 2: QKV projection.  C = xh @ Wt^T + b  (f16 in, f16 out)
// blockIdx.z selects Q/K/V.  Q,K stored (B,H,T,D); V stored transposed (B,H,D,T).
// 256 threads = 8 waves (4 on M x 2 on N); each wave: 2x4 16x16 sub-tiles.
// ---------------------------------------------------------------------------
__global__ __launch_bounds__(256) void proj_qkv_kernel(
    const _Float16* __restrict__ xh,
    const _Float16* __restrict__ Wqh, const float* __restrict__ bq,
    const _Float16* __restrict__ Wkh, const float* __restrict__ bk,
    const _Float16* __restrict__ Wvh, const float* __restrict__ bv,
    _Float16* __restrict__ Qb, _Float16* __restrict__ Kb, _Float16* __restrict__ Vt) {
    const int mode = blockIdx.z;
    const _Float16* __restrict__ Wt = (mode == 0) ? Wqh : (mode == 1) ? Wkh : Wvh;
    const float* __restrict__ bias  = (mode == 0) ? bq : (mode == 1) ? bk : bv;
    const int m0 = blockIdx.x * 128;
    const int n0 = blockIdx.y * 128;
    const int tid = threadIdx.x;
    const int lane = tid & 31, wid = tid >> 5;
    const int wm = wid & 3, wn = wid >> 2;
    const int half = lane >> 4, l16 = lane & 15;

    __shared__ __align__(32) _Float16 As[2][128][32]; // xh tiles (double buffered)
    __shared__ __align__(32) _Float16 Bs[2][128][32]; // Wt tiles [n][k]

    const v8f vzero = {};
    v8f acc[2][4];
#pragma unroll
    for (int ms = 0; ms < 2; ++ms)
#pragma unroll
        for (int ns = 0; ns < 4; ++ns) acc[ms][ns] = vzero;

    const _Float16* Ag = xh + (size_t)m0 * EDIM;
    const _Float16* Bg = Wt + (size_t)n0 * EDIM;
    GEMM_MAIN_LOOP(Ag, Bg);

    // epilogue: bias + scatter into head layouts
#pragma unroll
    for (int ns = 0; ns < 4; ++ns) {
        const int n = n0 + wn * 64 + ns * 16 + l16;
        const float bval = bias[n];
        const int h = n >> 6, d = n & 63;
#pragma unroll
        for (int ms = 0; ms < 2; ++ms) {
#pragma unroll
            for (int r = 0; r < 8; ++r) {
                const int m = m0 + wm * 32 + ms * 16 + r + 8 * half;
                const int b = m >> 11, t = m & 2047; // T = 2048
                const float val = acc[ms][ns][r] + bval;
                if (mode == 2) {
                    Vt[(size_t)((b * HHEADS + h) * DHEAD + d) * TSEQ + t] = (_Float16)val;
                } else if (mode == 0) {
                    Qb[(size_t)((b * HHEADS + h) * TSEQ + t) * DHEAD + d] = (_Float16)val;
                } else {
                    Kb[(size_t)((b * HHEADS + h) * TSEQ + t) * DHEAD + d] = (_Float16)val;
                }
            }
        }
    }
}

// ---------------------------------------------------------------------------
// Kernel 3: flash attention.  One block = 128 query rows of one (b,h).
// 8 independent waves, each owns 16 rows; Q lives in registers; online softmax.
// ---------------------------------------------------------------------------
__global__ __launch_bounds__(256) void flash_attn_kernel(
    const _Float16* __restrict__ Qb, const _Float16* __restrict__ Kb,
    const _Float16* __restrict__ Vt, const unsigned char* __restrict__ mask,
    const float* __restrict__ physb, _Float16* __restrict__ attn) {
    const int tid = threadIdx.x, lane = tid & 31, wid = tid >> 5;
    const int half = lane >> 4, l16 = lane & 15;
    const int nqb = TSEQ / 128;
    const int qb = blockIdx.x % nqb;
    const int bh = blockIdx.x / nqb;
    const int b = bh / HHEADS, h = bh % HHEADS;
    const float pbias = physb[bh];
    const float scale = 0.125f; // 1/sqrt(64)
    const size_t headQK = (size_t)bh * TSEQ * DHEAD;

    __shared__ __align__(32) _Float16 Pb[8][16][72]; // per-wave P staging (C->A layout)

    const int i0 = qb * 128 + wid * 16;
    const _Float16* qrow = Qb + headQK + (size_t)(i0 + l16) * DHEAD;
    v16h qa[2];
#pragma unroll
    for (int c = 0; c < 2; ++c)
        qa[c] = frag_a_2x8(qrow + c * 32 + half * 16, qrow + c * 32 + half * 16 + 8);

    const v8f vzero = {};
    v8f o[4];
#pragma unroll
    for (int n = 0; n < 4; ++n) o[n] = vzero;
    float mrow[8], lrow[8];
#pragma unroll
    for (int r = 0; r < 8; ++r) { mrow[r] = -1e30f; lrow[r] = 0.0f; }

    for (int j0 = 0; j0 < TSEQ; j0 += 64) {
        // S = Q @ K^T for 4 column tiles (16 keys each)
        v8f sc[4];
#pragma unroll
        for (int c2 = 0; c2 < 4; ++c2) {
            sc[c2] = vzero;
            const _Float16* krow = Kb + headQK + (size_t)(j0 + c2 * 16 + l16) * DHEAD;
#pragma unroll
            for (int c = 0; c < 2; ++c) {
                const v16h kf = *(const v16h*)(krow + c * 32 + half * 16);
                sc[c2] = __builtin_amdgcn_wmma_f32_16x16x32_f16(
                    false, qa[c], false, kf, (short)0, sc[c2], false, false);
            }
        }
        // scale + physics bias + key-padding mask (lane's 8 values share one key col)
#pragma unroll
        for (int c2 = 0; c2 < 4; ++c2) {
            const int j = j0 + c2 * 16 + l16;
            const bool keep = mask[(size_t)b * TSEQ + j] != 0;
#pragma unroll
            for (int r = 0; r < 8; ++r) {
                const float s = sc[c2][r] * scale + pbias;
                sc[c2][r] = keep ? s : -1e30f;
            }
        }
        // online softmax: row max via half-wave shfl reduction
        float newm[8], alpha[8], psum[8];
#pragma unroll
        for (int r = 0; r < 8; ++r) {
            float mx = fmaxf(fmaxf(sc[0][r], sc[1][r]), fmaxf(sc[2][r], sc[3][r]));
#pragma unroll
            for (int off = 1; off < 16; off <<= 1)
                mx = fmaxf(mx, __shfl_xor(mx, off, 32));
            newm[r] = fmaxf(mrow[r], mx);
            alpha[r] = __expf(mrow[r] - newm[r]);
            mrow[r] = newm[r];
            psum[r] = 0.0f;
        }
        // exp, accumulate row-sums, stage P (C-layout -> LDS -> A-layout)
#pragma unroll
        for (int c2 = 0; c2 < 4; ++c2) {
#pragma unroll
            for (int r = 0; r < 8; ++r) {
                const float p = __expf(sc[c2][r] - newm[r]);
                psum[r] += p;
                Pb[wid][r + 8 * half][c2 * 16 + l16] = (_Float16)p;
            }
        }
#pragma unroll
        for (int r = 0; r < 8; ++r) {
            float s = psum[r];
#pragma unroll
            for (int off = 1; off < 16; off <<= 1) s += __shfl_xor(s, off, 32);
            lrow[r] = lrow[r] * alpha[r] + s;
#pragma unroll
            for (int n = 0; n < 4; ++n) o[n][r] *= alpha[r];
        }
        // O += P @ V   (same-wave LDS store->load is in-order; no barrier needed)
#pragma unroll
        for (int c = 0; c < 2; ++c) {
            const _Float16* prow = &Pb[wid][l16][c * 32 + half * 16];
            const v16h pf = frag_a_2x8(prow, prow + 8);
#pragma unroll
            for (int n = 0; n < 4; ++n) {
                const _Float16* vrow =
                    Vt + (size_t)(bh * DHEAD + n * 16 + l16) * TSEQ + j0 + c * 32 + half * 16;
                const v16h vf = *(const v16h*)vrow;
                o[n] = __builtin_amdgcn_wmma_f32_16x16x32_f16(
                    false, pf, false, vf, (short)0, o[n], false, false);
            }
        }
    }
    // normalize and write (B,T,E) f16
#pragma unroll
    for (int n = 0; n < 4; ++n) {
#pragma unroll
        for (int r = 0; r < 8; ++r) {
            const int i = i0 + r + 8 * half;
            const int d = n * 16 + l16;
            const float val = o[n][r] / lrow[r];
            attn[((size_t)(b * TSEQ + i) * HHEADS + h) * DHEAD + d] = (_Float16)val;
        }
    }
}

// ---------------------------------------------------------------------------
// Kernel 4: output projection.  out = attn(f16) @ Woh^T + bo  -> fp32
// ---------------------------------------------------------------------------
__global__ __launch_bounds__(256) void out_proj_kernel(
    const _Float16* __restrict__ A, const _Float16* __restrict__ Woh,
    const float* __restrict__ bo, float* __restrict__ out) {
    const int m0 = blockIdx.x * 128;
    const int n0 = blockIdx.y * 128;
    const int tid = threadIdx.x;
    const int lane = tid & 31, wid = tid >> 5;
    const int wm = wid & 3, wn = wid >> 2;
    const int half = lane >> 4, l16 = lane & 15;

    __shared__ __align__(32) _Float16 As[2][128][32];
    __shared__ __align__(32) _Float16 Bs[2][128][32];

    const v8f vzero = {};
    v8f acc[2][4];
#pragma unroll
    for (int ms = 0; ms < 2; ++ms)
#pragma unroll
        for (int ns = 0; ns < 4; ++ns) acc[ms][ns] = vzero;

    const _Float16* Ag = A + (size_t)m0 * EDIM;
    const _Float16* Bg = Woh + (size_t)n0 * EDIM;
    GEMM_MAIN_LOOP(Ag, Bg);

#pragma unroll
    for (int ns = 0; ns < 4; ++ns) {
        const int n = n0 + wn * 64 + ns * 16 + l16;
        const float bval = bo[n];
#pragma unroll
        for (int ms = 0; ms < 2; ++ms) {
#pragma unroll
            for (int r = 0; r < 8; ++r) {
                const int m = m0 + wm * 32 + ms * 16 + r + 8 * half;
                out[(size_t)m * EDIM + n] = acc[ms][ns][r] + bval;
            }
        }
    }
}

// ---------------------------------------------------------------------------
extern "C" void kernel_launch(void* const* d_in, const int* in_sizes, int n_in,
                              void* d_out, int out_size, void* d_ws, size_t ws_size,
                              hipStream_t stream) {
    const float* x   = (const float*)d_in[0];
    const float* ps  = (const float*)d_in[1];
    const unsigned char* mask = (const unsigned char*)d_in[2];
    const float* Wq  = (const float*)d_in[3];
    const float* bq  = (const float*)d_in[4];
    const float* Wk  = (const float*)d_in[5];
    const float* bk  = (const float*)d_in[6];
    const float* Wv  = (const float*)d_in[7];
    const float* bv  = (const float*)d_in[8];
    const float* Wo  = (const float*)d_in[9];
    const float* bo  = (const float*)d_in[10];
    const float* Wp1 = (const float*)d_in[11];
    const float* bp1 = (const float*)d_in[12];
    const float* Wp2 = (const float*)d_in[13];
    const float* bp2 = (const float*)d_in[14];
    float* out = (float*)d_out;

    char* ws = (char*)d_ws;
    float* phys = (float*)ws;                       // 32 floats
    _Float16* Qb = (_Float16*)(ws + 256);
    const size_t hsz = (size_t)BATCH * HHEADS * TSEQ * DHEAD; // 4M elems
    const size_t wsz = (size_t)EDIM * EDIM;                   // 1M elems
    _Float16* Kb  = Qb + hsz;                       // +8MB
    _Float16* Vt  = Kb + hsz;                       // +8MB (stored B,H,D,T)
    _Float16* At  = Vt + hsz;                       // +8MB (attn out, B,T,E)
    _Float16* Xh  = At + hsz;                       // +8MB (x in f16)
    _Float16* Wqh = Xh + hsz;                       // +2MB each, transposed [n][k]
    _Float16* Wkh = Wqh + wsz;
    _Float16* Wvh = Wkh + wsz;
    _Float16* Woh = Wvh + wsz;

    physics_bias_kernel<<<1, 32, 0, stream>>>(ps, Wp1, bp1, Wp2, bp2, phys);
    convert_x_kernel<<<dim3((unsigned)(hsz / (256 * 8))), 256, 0, stream>>>(x, Xh);
    convert_w_kernel<<<dim3((unsigned)(wsz / 256), 4), 256, 0, stream>>>(
        Wq, Wk, Wv, Wo, Wqh, Wkh, Wvh, Woh);
    proj_qkv_kernel<<<dim3((BATCH * TSEQ) / 128, EDIM / 128, 3), 256, 0, stream>>>(
        Xh, Wqh, bq, Wkh, bk, Wvh, bv, Qb, Kb, Vt);
    flash_attn_kernel<<<dim3((TSEQ / 128) * BATCH * HHEADS), 256, 0, stream>>>(
        Qb, Kb, Vt, mask, phys, At);
    out_proj_kernel<<<dim3((BATCH * TSEQ) / 128, EDIM / 128), 256, 0, stream>>>(
        At, Woh, bo, out);
}
